// AssignmentSimilarityNet_26731876451184
// MI455X (gfx1250) — matrix operand order repaired and checked by reading, not compiled, not run on hardware
//
#include <hip/hip_runtime.h>
#include <math.h>

typedef __bf16 bf16_t;
typedef bf16_t v16bf __attribute__((ext_vector_type(16)));
typedef float  v8f   __attribute__((ext_vector_type(8)));
typedef unsigned short u16;
typedef unsigned int   u32;

#define AA 512
#define BBV 512
#define REIDV 512
#define NDV 128

union Frag { v16bf v; uint4 q[2]; u16 s[16]; };

__device__ __forceinline__ u16 f2bf(float f) {
    u32 u = __float_as_uint(f);
    u32 r = u + 0x7FFFu + ((u >> 16) & 1u);
    return (u16)(r >> 16);
}
__device__ __forceinline__ float bf2f(u16 s) { return __uint_as_float(((u32)s) << 16); }

__device__ __forceinline__ v8f wmma_bf16(const Frag& a, const Frag& b, v8f c) {
    return __builtin_amdgcn_wmma_f32_16x16x32_bf16(false, a.v, false, b.v, (short)0, c, false, false);
}

// A-matrix 16x32 bf16 frag: lane L -> row M = L&15; elems 0..7 K=k0+8*hl+e, elems 8..15 K=k0+16+8*hl+e
__device__ __forceinline__ Frag ldA(const u16* base, int ld, int m0, int k0, int l15, int hl) {
    const u16* p = base + (size_t)(m0 + l15) * ld + k0 + 8 * hl;
    Frag f;
    f.q[0] = *(const uint4*)p;
    f.q[1] = *(const uint4*)(p + 16);
    return f;
}
// B-matrix 32x16 bf16 frag from N-major (transposed) weights: lane L -> col N = n0+(L&15); elems 0..7 K=k0+8*hl+e, elems 8..15 K=k0+16+8*hl+e
__device__ __forceinline__ Frag ldB(const u16* baseT, int ld, int n0, int k0, int l15, int hl) {
    const u16* p = baseT + (size_t)(n0 + l15) * ld + k0 + 8 * hl;
    Frag f;
    f.q[0] = *(const uint4*)p;
    f.q[1] = *(const uint4*)(p + 16);
    return f;
}

// ---------------- prep: row-normalize apps + bf16 casts -----------------
__global__ void k_prep_vecs(const float* __restrict__ ta, const float* __restrict__ ca,
                            u16* anb, u16* bnb, u16* tappb, u16* cappb) {
    __shared__ float red[256];
    int row = blockIdx.x, t = threadIdx.x;
    const float* src; u16 *nrm, *raw;
    if (row < AA) { src = ta + (size_t)row * REIDV; nrm = anb + (size_t)row * REIDV; raw = tappb + (size_t)row * REIDV; }
    else { int r = row - AA; src = ca + (size_t)r * REIDV; nrm = bnb + (size_t)r * REIDV; raw = cappb + (size_t)r * REIDV; }
    float x0 = src[t], x1 = src[t + 256];
    red[t] = x0 * x0 + x1 * x1;
    __syncthreads();
    for (int o = 128; o > 0; o >>= 1) { if (t < o) red[t] += red[t + o]; __syncthreads(); }
    float rn = rsqrtf(red[0]);
    raw[t] = f2bf(x0); raw[t + 256] = f2bf(x1);
    nrm[t] = f2bf(x0 * rn); nrm[t + 256] = f2bf(x1 * rn);
}

// ---------------- prep: transpose + cast weights to bf16 ----------------
__global__ void k_prep_w(const float* __restrict__ Wcnn, const float* __restrict__ We1,
                         const float* __restrict__ We2, const float* __restrict__ Wc1,
                         const float* __restrict__ Wei2,
                         u16* WcnnT, u16* We1cT, u16* We1dT, u16* We2T, u16* Wc1T, u16* Wei2T) {
    int id = blockIdx.x * blockDim.x + threadIdx.x;
    if (id < REIDV * NDV) { int k = id / NDV, n = id % NDV; WcnnT[(size_t)n * REIDV + k] = f2bf(Wcnn[id]); return; }
    id -= REIDV * NDV;
    if (id < 4096) { int k = id / 64, n = id % 64; We1cT[n * 64 + k] = f2bf(We1[(256 + k) * 64 + n]); return; }
    id -= 4096;
    if (id < 4096) { int k = id / 64, n = id % 64; We1dT[n * 64 + k] = f2bf(We1[(320 + k) * 64 + n]); return; }
    id -= 4096;
    if (id < 4096) { int k = id / 64, n = id % 64; We2T[n * 64 + k] = f2bf(We2[k * 64 + n]); return; }
    id -= 4096;
    if (id < 4096) { int k = id / 64, n = id % 64; Wc1T[n * 64 + k] = f2bf(Wc1[k * 64 + n]); return; }
    id -= 4096;
    if (id < 4096) { int k = id / 64, n = id % 64; Wei2T[n * 64 + k] = f2bf(Wei2[k * 64 + n]); }
}

// ---------------- cos_dist = 1 - an @ bn^T (K=512, WMMA) ----------------
__global__ void __launch_bounds__(256) k_cos(const u16* __restrict__ anb, const u16* __restrict__ bnb,
                                             float* __restrict__ cosd) {
    int lane = threadIdx.x & 31, wid = threadIdx.x >> 5;
    int tile = blockIdx.x * 8 + wid;          // 1024 tiles of 16x16
    int a0 = (tile >> 5) << 4, b0 = (tile & 31) << 4;
    int l15 = lane & 15, hl = lane >> 4;
    v8f acc = {};
    for (int k0 = 0; k0 < REIDV; k0 += 32) {
        Frag fa = ldA(anb, REIDV, a0, k0, l15, hl);
        Frag fb = ldB(bnb, REIDV, b0, k0, l15, hl);   // bn rows are K-contiguous == N-major
        acc = wmma_bf16(fa, fb, acc);
    }
    for (int r = 0; r < 8; r++)
        cosd[(size_t)(a0 + r + 8 * hl) * BBV + b0 + l15] = 1.0f - acc[r];
}

// ---------------- na/nb = relu(app @ W_cnn + b) (K=512, WMMA) -----------
__global__ void __launch_bounds__(256) k_cnn(const u16* __restrict__ tappb, const u16* __restrict__ cappb,
                                             const u16* __restrict__ WcnnT, const float* __restrict__ bcnn,
                                             float* __restrict__ na, float* __restrict__ nb) {
    int lane = threadIdx.x & 31, wid = threadIdx.x >> 5;
    int tile = blockIdx.x * 8 + wid;          // 512 tiles (256 track + 256 current)
    const u16* src = (tile < 256) ? tappb : cappb;
    float* dst = (tile < 256) ? na : nb;
    int t = tile & 255;
    int m0 = (t >> 3) << 4, n0 = (t & 7) << 4;
    int l15 = lane & 15, hl = lane >> 4;
    v8f acc = {};
    for (int k0 = 0; k0 < REIDV; k0 += 32) {
        Frag fa = ldA(src, REIDV, m0, k0, l15, hl);
        Frag fb = ldB(WcnnT, REIDV, n0, k0, l15, hl);
        acc = wmma_bf16(fa, fb, acc);
    }
    float bias = bcnn[n0 + l15];
    for (int r = 0; r < 8; r++) {
        float v = acc[r] + bias;
        dst[(size_t)(m0 + r + 8 * hl) * NDV + n0 + l15] = v > 0.f ? v : 0.f;
    }
}

// ------- naC = na@W_e1[0:128]+b_e1 ; nbC = nb@W_e1[128:256] (small) ------
__global__ void k_naC(const float* __restrict__ na, const float* __restrict__ nb,
                      const float* __restrict__ We1, const float* __restrict__ be1,
                      float* __restrict__ naC, float* __restrict__ nbC) {
    int id = blockIdx.x * blockDim.x + threadIdx.x;   // 65536
    int n = id & 63, row = (id >> 6) & 511;
    bool isB = id >= (512 * 64);
    const float* src = isB ? nb : na;
    const float* W = We1 + (isB ? 128 * 64 : 0);
    float s = isB ? 0.0f : be1[n];
    for (int k = 0; k < NDV; k++) s += src[row * NDV + k] * W[k * 64 + n];
    (isB ? nbC : naC)[row * 64 + n] = s;
}

// ---- edge0 = mlp2(edge_feats, W_ei1, W_ei2): VALU K=6 layer + WMMA -----
__global__ void __launch_bounds__(256) k_edgefeat(
        const float* __restrict__ tc, const float* __restrict__ cc,
        const float* __restrict__ tt, const float* __restrict__ ct,
        const float* __restrict__ cosd,
        const float* __restrict__ Wei1, const float* __restrict__ bei1,
        const u16* __restrict__ Wei2T, const float* __restrict__ bei2,
        u16* __restrict__ edge, u16* __restrict__ iedge) {
    __shared__ __align__(16) u16 smem[8][16 * 64];
    int lane = threadIdx.x & 31, wid = threadIdx.x >> 5;
    int tile = blockIdx.x * 8 + wid;          // 16384 tiles
    int a = tile >> 5, b0 = (tile & 31) << 4;
    int l15 = lane & 15, hl = lane >> 4;
    int b = b0 + l15;
    float th = tc[a * 4 + 3] - tc[a * 4 + 1], tw = tc[a * 4 + 2] - tc[a * 4 + 0];
    float ch = cc[b * 4 + 3] - cc[b * 4 + 1], cw = cc[b * 4 + 2] - cc[b * 4 + 0];
    float xd = (cc[b * 4 + 0] + floorf(cw * 0.5f)) - (tc[a * 4 + 0] + floorf(tw * 0.5f));
    float yd = (cc[b * 4 + 1] + floorf(ch * 0.5f)) - (tc[a * 4 + 1] + floorf(th * 0.5f));
    float den = th + ch;
    float f[6];
    f[0] = 2.f * xd / den; f[1] = 2.f * yd / den;
    f[2] = __logf(th / ch); f[3] = __logf(tw / cw);
    f[4] = ct[b] - tt[a];  f[5] = cosd[(size_t)a * BBV + b];
    u16* lds = smem[wid];
    int nbase = hl * 32;                       // lane's 32 hidden outputs, pair = l15
    for (int n = 0; n < 32; n++) {
        int nn = nbase + n;
        float h = bei1[nn];
        for (int k = 0; k < 6; k++) h += f[k] * Wei1[k * 64 + nn];
        lds[l15 * 64 + nn] = f2bf(h > 0.f ? h : 0.f);
    }
    Frag h0 = ldA(lds, 64, 0, 0, l15, hl), h1 = ldA(lds, 64, 0, 32, l15, hl);
    v8f acc[4];
    for (int nt = 0; nt < 4; nt++) {
        v8f c = {};
        c = wmma_bf16(h0, ldB(Wei2T, 64, nt * 16, 0, l15, hl), c);
        c = wmma_bf16(h1, ldB(Wei2T, 64, nt * 16, 32, l15, hl), c);
        acc[nt] = c;
    }
    for (int nt = 0; nt < 4; nt++) {
        float bias = bei2[nt * 16 + l15];
        for (int r = 0; r < 8; r++) {
            float v = acc[nt][r] + bias;
            lds[(r + 8 * hl) * 64 + nt * 16 + l15] = f2bf(v > 0.f ? v : 0.f);
        }
    }
    size_t base = ((size_t)a * BBV + b0) * 64;    // tile is contiguous 2KB
    const uint4* ls = (const uint4*)lds;
    uint4* e4 = (uint4*)(edge + base);
    uint4* i4 = (uint4*)(iedge + base);
    for (int i = 0; i < 4; i++) { uint4 d = ls[lane + i * 32]; e4[lane + i * 32] = d; i4[lane + i * 32] = d; }
}

// --------------- main fused message-passing edge step -------------------
__global__ void __launch_bounds__(256) k_edge(
        const u16* __restrict__ iedge, u16* __restrict__ edge,
        const float* __restrict__ naC, const float* __restrict__ nbC,
        const u16* __restrict__ We1cT, const u16* __restrict__ We1dT,
        const u16* __restrict__ We2T, const u16* __restrict__ Wc1T,
        const float* __restrict__ be2, const float* __restrict__ bc1,
        const float* __restrict__ wc2, const float* __restrict__ bc2,
        float* __restrict__ out) {
    __shared__ __align__(16) u16 smem[8][16 * 64];
    int lane = threadIdx.x & 31, wid = threadIdx.x >> 5;
    int tile = blockIdx.x * 8 + wid;          // 16384 tiles
    int a = tile >> 5, b0 = (tile & 31) << 4;
    int l15 = lane & 15, hl = lane >> 4;
    size_t base = ((size_t)a * BBV + b0) * 64;
    const u16* eT = edge + base;
    const u16* iT = iedge + base;
    Frag e0 = ldA(eT, 64, 0, 0, l15, hl), e1 = ldA(eT, 64, 0, 32, l15, hl);
    Frag i0 = ldA(iT, 64, 0, 0, l15, hl), i1 = ldA(iT, 64, 0, 32, l15, hl);
    u16* lds = smem[wid];
    v8f acc[4];
    // h1 = relu(edge@We1c + init@We1d + naC[a] + nbC[b])   (b_e1 folded into naC)
    for (int nt = 0; nt < 4; nt++) {
        v8f c = {};
        c = wmma_bf16(e0, ldB(We1cT, 64, nt * 16, 0, l15, hl), c);
        c = wmma_bf16(e1, ldB(We1cT, 64, nt * 16, 32, l15, hl), c);
        c = wmma_bf16(i0, ldB(We1dT, 64, nt * 16, 0, l15, hl), c);
        c = wmma_bf16(i1, ldB(We1dT, 64, nt * 16, 32, l15, hl), c);
        acc[nt] = c;
    }
    float naCv[4];
    for (int nt = 0; nt < 4; nt++) naCv[nt] = naC[a * 64 + nt * 16 + l15];
    for (int r = 0; r < 8; r++) {
        int m = r + 8 * hl, brow = b0 + m;
        for (int nt = 0; nt < 4; nt++) {
            float v = acc[nt][r] + naCv[nt] + nbC[brow * 64 + nt * 16 + l15];
            lds[m * 64 + nt * 16 + l15] = f2bf(v > 0.f ? v : 0.f);
        }
    }
    // edge_new = relu(h1 @ We2 + b_e2)
    Frag h0 = ldA(lds, 64, 0, 0, l15, hl), h1f = ldA(lds, 64, 0, 32, l15, hl);
    for (int nt = 0; nt < 4; nt++) {
        v8f c = {};
        c = wmma_bf16(h0, ldB(We2T, 64, nt * 16, 0, l15, hl), c);
        c = wmma_bf16(h1f, ldB(We2T, 64, nt * 16, 32, l15, hl), c);
        acc[nt] = c;
    }
    for (int nt = 0; nt < 4; nt++) {
        float bias = be2[nt * 16 + l15];
        for (int r = 0; r < 8; r++) {
            float v = acc[nt][r] + bias;
            lds[(r + 8 * hl) * 64 + nt * 16 + l15] = f2bf(v > 0.f ? v : 0.f);
        }
    }
    {   // flush new edge tile (contiguous 2KB) to global, in place
        const uint4* ls = (const uint4*)lds;
        uint4* e4 = (uint4*)(edge + base);
        for (int i = 0; i < 4; i++) e4[lane + i * 32] = ls[lane + i * 32];
    }
    // logits = relu(edge_new @ Wc1 + b_c1) . wc2 + b_c2 ; sigmoid
    Frag c0 = ldA(lds, 64, 0, 0, l15, hl), c1 = ldA(lds, 64, 0, 32, l15, hl);
    for (int nt = 0; nt < 4; nt++) {
        v8f c = {};
        c = wmma_bf16(c0, ldB(Wc1T, 64, nt * 16, 0, l15, hl), c);
        c = wmma_bf16(c1, ldB(Wc1T, 64, nt * 16, 32, l15, hl), c);
        acc[nt] = c;
    }
    float wv[4], bc1v[4];
    for (int nt = 0; nt < 4; nt++) { wv[nt] = wc2[nt * 16 + l15]; bc1v[nt] = bc1[nt * 16 + l15]; }
    float bc2v = bc2[0];
    for (int r = 0; r < 8; r++) {
        float p = 0.f;
        for (int nt = 0; nt < 4; nt++) {
            float v = acc[nt][r] + bc1v[nt];
            v = v > 0.f ? v : 0.f;
            p += v * wv[nt];
        }
        p += __shfl_xor(p, 1); p += __shfl_xor(p, 2); p += __shfl_xor(p, 4); p += __shfl_xor(p, 8);
        if (l15 == 0) {
            float lg = p + bc2v;
            out[(size_t)a * BBV + b0 + r + 8 * hl] = 1.f / (1.f + __expf(-lg));
        }
    }
}

// ---------------- edge.sum(axis=1) and edge.sum(axis=0) -----------------
__global__ void k_sums(const u16* __restrict__ edge, float* __restrict__ rowsum, float* __restrict__ colsum) {
    int idx = blockIdx.x, t = threadIdx.x;    // 64 threads
    if (idx < AA) {
        const u16* p = edge + (size_t)idx * BBV * 64 + t;
        float s0 = 0, s1 = 0, s2 = 0, s3 = 0;
        for (int b = 0; b < BBV; b += 4) {
            s0 += bf2f(p[(size_t)b * 64]);       s1 += bf2f(p[(size_t)(b + 1) * 64]);
            s2 += bf2f(p[(size_t)(b + 2) * 64]); s3 += bf2f(p[(size_t)(b + 3) * 64]);
        }
        rowsum[idx * 64 + t] = (s0 + s1) + (s2 + s3);
    } else {
        int b = idx - AA;
        const u16* p = edge + (size_t)b * 64 + t;
        float s0 = 0, s1 = 0, s2 = 0, s3 = 0;
        for (int a = 0; a < AA; a += 4) {
            s0 += bf2f(p[(size_t)a * BBV * 64]);       s1 += bf2f(p[(size_t)(a + 1) * BBV * 64]);
            s2 += bf2f(p[(size_t)(a + 2) * BBV * 64]); s3 += bf2f(p[(size_t)(a + 3) * BBV * 64]);
        }
        colsum[b * 64 + t] = (s0 + s1) + (s2 + s3);
    }
}

// ---------------- node MLP: na/nb = mlp2([n, sum], Wn1, Wn2) ------------
__global__ void k_node(const float* __restrict__ naCur, const float* __restrict__ nbCur,
                       const float* __restrict__ rowsum, const float* __restrict__ colsum,
                       const float* __restrict__ Wn1, const float* __restrict__ bn1,
                       const float* __restrict__ Wn2, const float* __restrict__ bn2,
                       float* __restrict__ naNxt, float* __restrict__ nbNxt) {
    __shared__ float x[192];
    __shared__ float h[128];
    int row = blockIdx.x, t = threadIdx.x;    // 128 threads
    bool isA = row < AA;
    int r = isA ? row : row - AA;
    const float* src = isA ? naCur : nbCur;
    const float* sm = isA ? rowsum : colsum;
    x[t] = src[r * NDV + t];
    if (t < 64) x[128 + t] = sm[r * 64 + t];
    __syncthreads();
    float s = bn1[t];
    for (int k = 0; k < 192; k++) s += x[k] * Wn1[k * NDV + t];
    h[t] = s > 0.f ? s : 0.f;
    __syncthreads();
    float o = bn2[t];
    for (int k = 0; k < 128; k++) o += h[k] * Wn2[k * NDV + t];
    (isA ? naNxt : nbNxt)[r * NDV + t] = o > 0.f ? o : 0.f;
}

extern "C" void kernel_launch(void* const* d_in, const int* in_sizes, int n_in,
                              void* d_out, int out_size, void* d_ws, size_t ws_size,
                              hipStream_t stream) {
    const float* track_app   = (const float*)d_in[0];
    const float* current_app = (const float*)d_in[1];
    const float* track_c     = (const float*)d_in[2];
    const float* current_c   = (const float*)d_in[3];
    const float* track_t     = (const float*)d_in[4];
    const float* curr_t      = (const float*)d_in[5];
    const float* Wcnn = (const float*)d_in[6];  const float* bcnn = (const float*)d_in[7];
    const float* Wei1 = (const float*)d_in[8];  const float* bei1 = (const float*)d_in[9];
    const float* Wei2 = (const float*)d_in[10]; const float* bei2 = (const float*)d_in[11];
    const float* We1  = (const float*)d_in[12]; const float* be1  = (const float*)d_in[13];
    const float* We2  = (const float*)d_in[14]; const float* be2  = (const float*)d_in[15];
    const float* Wn1  = (const float*)d_in[16]; const float* bn1  = (const float*)d_in[17];
    const float* Wn2  = (const float*)d_in[18]; const float* bn2  = (const float*)d_in[19];
    const float* Wc1  = (const float*)d_in[20]; const float* bc1  = (const float*)d_in[21];
    const float* Wc2  = (const float*)d_in[22]; const float* bc2  = (const float*)d_in[23];
    float* out = (float*)d_out;

    char* w = (char*)d_ws;
    auto alloc = [&](size_t bytes) -> char* {
        char* p = w;
        w += (bytes + 255) & ~(size_t)255;
        return p;
    };
    u16* anb    = (u16*)alloc(AA * REIDV * 2);
    u16* bnb    = (u16*)alloc(BBV * REIDV * 2);
    u16* tappb  = (u16*)alloc(AA * REIDV * 2);
    u16* cappb  = (u16*)alloc(BBV * REIDV * 2);
    u16* WcnnT  = (u16*)alloc(REIDV * NDV * 2);
    u16* We1cT  = (u16*)alloc(4096 * 2);
    u16* We1dT  = (u16*)alloc(4096 * 2);
    u16* We2T   = (u16*)alloc(4096 * 2);
    u16* Wc1T   = (u16*)alloc(4096 * 2);
    u16* Wei2T  = (u16*)alloc(4096 * 2);
    float* cosd = (float*)alloc((size_t)AA * BBV * 4);
    u16* edge   = (u16*)alloc((size_t)AA * BBV * 64 * 2);
    u16* iedge  = (u16*)alloc((size_t)AA * BBV * 64 * 2);
    float* na0  = (float*)alloc(AA * NDV * 4);
    float* nb0  = (float*)alloc(BBV * NDV * 4);
    float* na1  = (float*)alloc(AA * NDV * 4);
    float* nb1  = (float*)alloc(BBV * NDV * 4);
    float* naC  = (float*)alloc(AA * 64 * 4);
    float* nbC  = (float*)alloc(BBV * 64 * 4);
    float* rsum = (float*)alloc(AA * 64 * 4);
    float* csum = (float*)alloc(BBV * 64 * 4);

    k_prep_vecs<<<1024, 256, 0, stream>>>(track_app, current_app, anb, bnb, tappb, cappb);
    k_prep_w<<<(REIDV * NDV + 5 * 4096 + 255) / 256, 256, 0, stream>>>(
        Wcnn, We1, We2, Wc1, Wei2, WcnnT, We1cT, We1dT, We2T, Wc1T, Wei2T);
    k_cos<<<128, 256, 0, stream>>>(anb, bnb, cosd);
    k_cnn<<<64, 256, 0, stream>>>(tappb, cappb, WcnnT, bcnn, na0, nb0);
    k_edgefeat<<<2048, 256, 0, stream>>>(track_c, current_c, track_t, curr_t, cosd,
                                         Wei1, bei1, Wei2T, bei2, edge, iedge);
    k_naC<<<256, 256, 0, stream>>>(na0, nb0, We1, be1, naC, nbC);

    float *naCur = na0, *nbCur = nb0, *naNxt = na1, *nbNxt = nb1;
    for (int s = 0; s < 4; s++) {
        k_edge<<<2048, 256, 0, stream>>>(iedge, edge, naC, nbC, We1cT, We1dT, We2T, Wc1T,
                                         be2, bc1, Wc2, bc2, out + (size_t)s * AA * BBV);
        if (s < 3) {
            k_sums<<<1024, 64, 0, stream>>>(edge, rsum, csum);
            k_node<<<1024, 128, 0, stream>>>(naCur, nbCur, rsum, csum, Wn1, bn1, Wn2, bn2, naNxt, nbNxt);
            float* t0 = naCur; naCur = naNxt; naNxt = t0;
            float* t1 = nbCur; nbCur = nbNxt; nbNxt = t1;
            k_naC<<<256, 256, 0, stream>>>(naCur, nbCur, We1, be1, naC, nbC);
        }
    }
}
